// Model_39393440039361
// MI455X (gfx1250) — matrix operand order repaired and checked by reading, not compiled, not run on hardware
//
#include <hip/hip_runtime.h>
#include <hip/hip_bf16.h>

#define N_NODES 50000
#define N_EDGES 600000
#define HIDDEN  128
#define EDGE_DIM 64
#define OUT_DIM  16

typedef __attribute__((ext_vector_type(16))) __bf16 v16bf;
typedef __attribute__((ext_vector_type(8)))  float  v8f;

// ---------------------------------------------------------------------------
// helpers
// ---------------------------------------------------------------------------
__device__ __forceinline__ float bfbits_to_f32(unsigned short u) {
    union { unsigned int i; float f; } c;
    c.i = ((unsigned int)u) << 16;
    return c.f;
}

// ---------------------------------------------------------------------------
// Weight pre-pack: f32 [K][NC] row-major -> bf16 WMMA B-fragment order
// packed index: ((kt*NT + nt)*32 + lane)*16 + i
//   lane: n = nt*16 + (lane&15), K range = kt*32 + (lane>>4)*16 + i
// ---------------------------------------------------------------------------
template <int KT, int NT>
__global__ void pack_b_kernel(const float* __restrict__ W, __bf16* __restrict__ P) {
    constexpr int NC    = NT * 16;
    constexpr int TOTAL = KT * NT * 32 * 16;
    int idx = blockIdx.x * blockDim.x + threadIdx.x;
    if (idx >= TOTAL) return;
    int i    = idx & 15;
    int lane = (idx >> 4) & 31;
    int f    = idx >> 9;
    int kt   = f / NT;
    int nt   = f % NT;
    int k = kt * 32 + (lane >> 4) * 16 + i;
    int n = nt * 16 + (lane & 15);
    P[idx] = (__bf16)W[(size_t)k * NC + n];
}

// ---------------------------------------------------------------------------
// Generic bf16-WMMA GEMM:  Y = act((A [+ A2]) @ W + bias)
// K = KT*32, out cols = NT*16. One wave per 16-row M tile.
// Packed B staged in LDS once per block; A loaded as aligned float4 pairs.
// ---------------------------------------------------------------------------
template <int KT, int NT, bool RELU, bool ADD2, bool BF16OUT>
__global__ void gemm_kernel(const float* __restrict__ A,
                            const float* __restrict__ A2,
                            const __bf16* __restrict__ Bp,
                            const float* __restrict__ bias,
                            void* __restrict__ Yv, int M) {
    constexpr int K     = KT * 32;
    constexpr int NC    = NT * 16;
    constexpr int NFRAG = KT * NT;

    __shared__ __bf16 Blds[NFRAG * 512];

    // cooperative stage of packed weights (all waves participate, pre-exit)
    {
        const uint4* s = (const uint4*)Bp;
        uint4*       d = (uint4*)Blds;
        for (int i = threadIdx.x; i < NFRAG * 64; i += blockDim.x) d[i] = s[i];
    }
    __syncthreads();

    const int lane  = threadIdx.x & 31;
    const int wave  = blockIdx.x * (blockDim.x >> 5) + (threadIdx.x >> 5);
    const int mbase = wave * 16;
    if (mbase >= M) return;                 // wave-uniform: EXEC all-1s below
    const int hl = lane >> 4;
    const int mn = lane & 15;

    v8f acc[NT];
    #pragma unroll
    for (int nt = 0; nt < NT; ++nt)
        #pragma unroll
        for (int r = 0; r < 8; ++r) acc[nt][r] = 0.0f;

    const float* rowA  = A + (size_t)(mbase + mn) * K;
    const float* rowA2 = ADD2 ? (A2 + (size_t)(mbase + mn) * K) : rowA;

    #pragma unroll
    for (int kt = 0; kt < KT; ++kt) {
        const int kb = kt * 32 + hl * 16;   // lane's K base (64B aligned)
        float4 x0 = *(const float4*)(rowA + kb);
        float4 x1 = *(const float4*)(rowA + kb + 4);
        float4 y0 = *(const float4*)(rowA + kb + 8);
        float4 y1 = *(const float4*)(rowA + kb + 12);
        if (ADD2) {
            float4 u0 = *(const float4*)(rowA2 + kb);
            float4 u1 = *(const float4*)(rowA2 + kb + 4);
            float4 w0 = *(const float4*)(rowA2 + kb + 8);
            float4 w1 = *(const float4*)(rowA2 + kb + 12);
            x0.x += u0.x; x0.y += u0.y; x0.z += u0.z; x0.w += u0.w;
            x1.x += u1.x; x1.y += u1.y; x1.z += u1.z; x1.w += u1.w;
            y0.x += w0.x; y0.y += w0.y; y0.z += w0.z; y0.w += w0.w;
            y1.x += w1.x; y1.y += w1.y; y1.z += w1.z; y1.w += w1.w;
        }
        v16bf a;
        a[0]  = (__bf16)x0.x; a[1]  = (__bf16)x0.y; a[2]  = (__bf16)x0.z; a[3]  = (__bf16)x0.w;
        a[4]  = (__bf16)x1.x; a[5]  = (__bf16)x1.y; a[6]  = (__bf16)x1.z; a[7]  = (__bf16)x1.w;
        a[8]  = (__bf16)y0.x; a[9]  = (__bf16)y0.y; a[10] = (__bf16)y0.z; a[11] = (__bf16)y0.w;
        a[12] = (__bf16)y1.x; a[13] = (__bf16)y1.y; a[14] = (__bf16)y1.z; a[15] = (__bf16)y1.w;

        #pragma unroll
        for (int nt = 0; nt < NT; ++nt) {
            const v16bf b = *(const v16bf*)&Blds[((kt * NT + nt) * 32 + lane) * 16];
            acc[nt] = __builtin_amdgcn_wmma_f32_16x16x32_bf16(
                false, a, false, b, (short)0, acc[nt], false, false);
        }
    }

    #pragma unroll
    for (int nt = 0; nt < NT; ++nt) {
        float bv = bias[nt * 16 + mn];
        #pragma unroll
        for (int r = 0; r < 8; ++r) {
            float v = acc[nt][r] + bv;
            if (RELU) v = fmaxf(v, 0.0f);
            size_t oi = (size_t)(mbase + r + 8 * hl) * NC + nt * 16 + mn;
            if (BF16OUT) ((__bf16*)Yv)[oi] = (__bf16)v;
            else         ((float*)Yv)[oi]  = v;
        }
    }
}

// ---------------------------------------------------------------------------
// zero the aggregation buffer
// ---------------------------------------------------------------------------
__global__ void zero_kernel(float4* __restrict__ p, int n4) {
    int i = blockIdx.x * blockDim.x + threadIdx.x;
    if (i < n4) p[i] = make_float4(0.f, 0.f, 0.f, 0.f);
}

// ---------------------------------------------------------------------------
// fused gather + relu + scatter-add: 32 lanes per edge, 4 cols per lane
// ---------------------------------------------------------------------------
__global__ void message_scatter_kernel(const float* __restrict__ X,
                                       const __bf16* __restrict__ E,
                                       const int* __restrict__ src,
                                       const int* __restrict__ dst,
                                       float* __restrict__ aggr) {
    int t = blockIdx.x * blockDim.x + threadIdx.x;
    int eidx = t >> 5;
    if (eidx >= N_EDGES) return;
    int c = (t & 31) * 4;
    int s = src[eidx];
    int d = dst[eidx];

    const unsigned short* ep = (const unsigned short*)E + (size_t)eidx * HIDDEN + c;
    __builtin_prefetch(ep + 8 * HIDDEN, 0, 0);   // global_prefetch on the e stream

    float4  xv = *(const float4*)(X + (size_t)s * HIDDEN + c);
    ushort4 ev = *(const ushort4*)ep;

    float m0 = fmaxf(xv.x + bfbits_to_f32(ev.x), 0.0f);
    float m1 = fmaxf(xv.y + bfbits_to_f32(ev.y), 0.0f);
    float m2 = fmaxf(xv.z + bfbits_to_f32(ev.z), 0.0f);
    float m3 = fmaxf(xv.w + bfbits_to_f32(ev.w), 0.0f);

    float* ap = aggr + (size_t)d * HIDDEN + c;
    atomicAdd(ap + 0, m0);
    atomicAdd(ap + 1, m1);
    atomicAdd(ap + 2, m2);
    atomicAdd(ap + 3, m3);
}

// ---------------------------------------------------------------------------
// launch
// ---------------------------------------------------------------------------
extern "C" void kernel_launch(void* const* d_in, const int* in_sizes, int n_in,
                              void* d_out, int out_size, void* d_ws, size_t ws_size,
                              hipStream_t stream) {
    const float* x      = (const float*)d_in[0];
    const int*   eidx   = (const int*)  d_in[1];   // [2, E]
    const float* ea     = (const float*)d_in[2];
    const float* lin_w  = (const float*)d_in[3];
    const float* lin_b  = (const float*)d_in[4];
    const float* nn_w   = (const float*)d_in[5];
    const float* nn_b   = (const float*)d_in[6];
    const float* cls_w1 = (const float*)d_in[7];
    const float* cls_b1 = (const float*)d_in[8];
    const float* cls_w2 = (const float*)d_in[9];
    const float* cls_b2 = (const float*)d_in[10];
    const int* src = eidx;
    const int* dst = eidx + N_EDGES;

    // workspace layout
    char* ws = (char*)d_ws;
    size_t off = 0;
    __bf16* e_bf = (__bf16*)(ws + off); off += (size_t)N_EDGES * HIDDEN * sizeof(__bf16);
    float*  aggr = (float*)(ws + off);  off += (size_t)N_NODES * HIDDEN * sizeof(float);
    float*  xb0  = (float*)(ws + off);  off += (size_t)N_NODES * HIDDEN * sizeof(float);
    float*  xb1  = (float*)(ws + off);  off += (size_t)N_NODES * HIDDEN * sizeof(float);
    float*  hbuf = (float*)(ws + off);  off += (size_t)N_NODES * (HIDDEN / 4) * sizeof(float);
    __bf16* Pedge = (__bf16*)(ws + off); off += (size_t)2 * 8 * 512 * sizeof(__bf16);
    __bf16* Pnn   = (__bf16*)(ws + off); off += (size_t)4 * 8 * 512 * sizeof(__bf16);
    __bf16* Pc1   = (__bf16*)(ws + off); off += (size_t)4 * 2 * 512 * sizeof(__bf16);
    __bf16* Pc2   = (__bf16*)(ws + off); off += (size_t)1 * 1 * 512 * sizeof(__bf16);
    float*  outp  = (float*)d_out;

    const int WPB = 8;   // waves per block (256 threads)

    // 0) pack all weight matrices into WMMA B-fragment order (bf16)
    pack_b_kernel<2, 8><<<(2 * 8 * 512 + 255) / 256, 256, 0, stream>>>(lin_w,  Pedge);
    pack_b_kernel<4, 8><<<(4 * 8 * 512 + 255) / 256, 256, 0, stream>>>(nn_w,   Pnn);
    pack_b_kernel<4, 2><<<(4 * 2 * 512 + 255) / 256, 256, 0, stream>>>(cls_w1, Pc1);
    pack_b_kernel<1, 1><<<(1 * 1 * 512 + 255) / 256, 256, 0, stream>>>(cls_w2, Pc2);

    // 1) edge projection (loop-invariant): e = bf16(edge_attr @ W + b)
    {
        int tiles = N_EDGES / 16;
        int grid  = (tiles + WPB - 1) / WPB;
        gemm_kernel<2, 8, false, false, true><<<grid, 256, 0, stream>>>(
            ea, nullptr, Pedge, lin_b, e_bf, N_EDGES);
    }

    // 2) three GINE layers
    const float* xin = x;
    float* bufs[2] = {xb0, xb1};
    for (int layer = 0; layer < 3; ++layer) {
        int n4 = N_NODES * HIDDEN / 4;
        zero_kernel<<<(n4 + 255) / 256, 256, 0, stream>>>((float4*)aggr, n4);

        int nthreads = N_EDGES * 32;
        message_scatter_kernel<<<(nthreads + 255) / 256, 256, 0, stream>>>(
            xin, e_bf, src, dst, aggr);

        int tiles = N_NODES / 16;
        int grid  = (tiles + WPB - 1) / WPB;
        float* xout = bufs[layer & 1];
        if (layer < 2)
            gemm_kernel<4, 8, true,  true, false><<<grid, 256, 0, stream>>>(
                xin, aggr, Pnn, nn_b, xout, N_NODES);
        else
            gemm_kernel<4, 8, false, true, false><<<grid, 256, 0, stream>>>(
                xin, aggr, Pnn, nn_b, xout, N_NODES);
        xin = xout;
    }

    // 3) classifier MLP
    {
        int tiles = N_NODES / 16;
        int grid  = (tiles + WPB - 1) / WPB;
        gemm_kernel<4, 2, true,  false, false><<<grid, 256, 0, stream>>>(
            xin, nullptr, Pc1, cls_b1, hbuf, N_NODES);
        gemm_kernel<1, 1, false, false, false><<<grid, 256, 0, stream>>>(
            hbuf, nullptr, Pc2, cls_b2, outp, N_NODES);
    }
}
